// SelfAttention_49022756717131
// MI455X (gfx1250) — compile-verified, matrix operand order checked by reading, not run on hardware
//
#include <hip/hip_runtime.h>
#include <hip/hip_bf16.h>
#include <math.h>

// ---------------------------------------------------------------------------
// Types for CDNA5 WMMA (wave32)
// ---------------------------------------------------------------------------
typedef __bf16 bf16_t;
typedef __attribute__((ext_vector_type(16))) __bf16 v16bf;
typedef __attribute__((ext_vector_type(8)))  __bf16 v8bf;
typedef __attribute__((ext_vector_type(8)))  float  v8f;

// Vector-of-4-int type matching the async-LDS builtin's parameter type
typedef int i32x4 __attribute__((vector_size(16)));
typedef __attribute__((address_space(1))) i32x4* gptr_i32x4;
typedef __attribute__((address_space(3))) i32x4* lptr_i32x4;

#if __has_builtin(__builtin_amdgcn_global_load_async_to_lds_b128)
#define HAVE_ASYNC_LDS 1
#else
#define HAVE_ASYNC_LDS 0
#endif

static __device__ __forceinline__ bf16_t f2bf(float f) {
    union { float f; unsigned u; } in; in.f = f;
    unsigned u = in.u;
    u += 0x7FFFu + ((u >> 16) & 1u);           // round-to-nearest-even
    union { unsigned short s; bf16_t b; } out;
    out.s = (unsigned short)(u >> 16);
    return out.b;
}
static __device__ __forceinline__ float bf2f(bf16_t b) {
    union { unsigned short s; bf16_t b; } in; in.b = b;
    union { float f; unsigned u; } out; out.u = ((unsigned)in.s) << 16;
    return out.f;
}

// 16-byte global->LDS copy; async DMA path on CDNA5 if the builtin exists.
// as(1) pointers are bit-identical to generic; as(3) is the low 32 bits of a
// generic LDS pointer (ISA 10.2 aperture rules).
static __device__ __forceinline__ void cp16_g2l(const void* g, void* l) {
#if HAVE_ASYNC_LDS
    __builtin_amdgcn_global_load_async_to_lds_b128(
        (gptr_i32x4)(size_t)g,
        (lptr_i32x4)(unsigned)(size_t)l, 0, 0);
#else
    *(uint4*)l = *(const uint4*)g;
#endif
}
static __device__ __forceinline__ void cp_wait() {
#if HAVE_ASYNC_LDS
#if __has_builtin(__builtin_amdgcn_s_wait_asynccnt)
    __builtin_amdgcn_s_wait_asynccnt(0);
#else
    asm volatile("s_wait_asynccnt 0" ::: "memory");
#endif
#endif
}

// ---------------------------------------------------------------------------
// Fragment loader (CDNA5 16-bit A/B layout, ISA 7.12.2):
//   lane L holds row/col (L&15); lane-half kg=(L>>4) selects K-subgroups:
//   elements 0..7  -> K = k0 + kg*8 + e        (contiguous 8)
//   elements 8..15 -> K = k0 + 16 + kg*8 + e   (contiguous 8)
// ---------------------------------------------------------------------------
static __device__ __forceinline__ v16bf frag_ld(const bf16_t* rowptr, int kg, int k0) {
    v8bf lo = *(const v8bf*)(rowptr + k0 + kg * 8);
    v8bf hi = *(const v8bf*)(rowptr + k0 + 16 + kg * 8);
    v16bf f;
#pragma unroll
    for (int i = 0; i < 8; ++i) { f[i] = lo[i]; f[i + 8] = hi[i]; }
    return f;
}

static __device__ __forceinline__ float half_reduce_max(float v) {
#pragma unroll
    for (int off = 1; off < 16; off <<= 1) v = fmaxf(v, __shfl_xor(v, off, 32));
    return v;
}
static __device__ __forceinline__ float half_reduce_sum(float v) {
#pragma unroll
    for (int off = 1; off < 16; off <<= 1) v += __shfl_xor(v, off, 32);
    return v;
}

// ---------------------------------------------------------------------------
// GEMM: C[M,N] = A[M,K] * W[K,N], bf16 WMMA, fp32 accumulate.
// Block tile 128x128, K-step 32, 256 threads = 8 waves (4x2), each wave 32x64.
// Double-buffered LDS, register-staged global loads, one barrier per K-step.
// EPI==0: scatter qkv -> q[B,H,T,hd], k[B,H,T,hd], v TRANSPOSED [B,H,hd,T]
// EPI==1: plain fp32 row-major store
// ---------------------------------------------------------------------------
template <int EPI, bool A_BF16>
__global__ __launch_bounds__(256)
void gemm_wmma(const void* __restrict__ Aptr, const float* __restrict__ W,
               int M, int N, int K,
               bf16_t* __restrict__ qb, bf16_t* __restrict__ kb,
               bf16_t* __restrict__ vb, float* __restrict__ outp) {
    constexpr int BM = 128, BN = 128, BK = 32;
    __shared__ alignas(16) bf16_t As[2][BM][BK];   // [m][k]
    __shared__ alignas(16) bf16_t Bs[2][BN][BK];   // [n][k] (transposed on store)

    const int t = threadIdx.x;
    const int m0 = blockIdx.y * BM;
    const int n0 = blockIdx.x * BN;
    const int wave = t >> 5;
    const int lane = t & 31;
    const int l16 = lane & 15;
    const int kg = lane >> 4;
    const int wm = (wave >> 1) * 32;
    const int wn = (wave & 1) * 64;

    const float*  Af = (const float*)Aptr;
    const bf16_t* Ab = (const bf16_t*)Aptr;

    float4 ra[4], rb[4];
    v8bf   rab[2];

    auto ld_regs = [&](int k0) {
        if constexpr (!A_BF16) {
#pragma unroll
            for (int i = 0; i < 4; ++i) {
                int f = t + i * 256, r = f >> 3, c4 = f & 7;
                ra[i] = *(const float4*)(Af + (size_t)(m0 + r) * K + k0 + c4 * 4);
            }
        } else {
#pragma unroll
            for (int i = 0; i < 2; ++i) {
                int f = t + i * 256, r = f >> 2, c8 = f & 3;
                rab[i] = *(const v8bf*)(Ab + (size_t)(m0 + r) * K + k0 + c8 * 8);
            }
        }
#pragma unroll
        for (int i = 0; i < 4; ++i) {
            int f = t + i * 256, r = f >> 5, c4 = f & 31;
            rb[i] = *(const float4*)(W + (size_t)(k0 + r) * N + n0 + c4 * 4);
        }
    };
    auto st_tile = [&](int buf) {
        if constexpr (!A_BF16) {
#pragma unroll
            for (int i = 0; i < 4; ++i) {
                int f = t + i * 256, r = f >> 3, c4 = f & 7;
                As[buf][r][c4 * 4 + 0] = f2bf(ra[i].x);
                As[buf][r][c4 * 4 + 1] = f2bf(ra[i].y);
                As[buf][r][c4 * 4 + 2] = f2bf(ra[i].z);
                As[buf][r][c4 * 4 + 3] = f2bf(ra[i].w);
            }
        } else {
#pragma unroll
            for (int i = 0; i < 2; ++i) {
                int f = t + i * 256, r = f >> 2, c8 = f & 3;
                *(v8bf*)&As[buf][r][c8 * 8] = rab[i];
            }
        }
#pragma unroll
        for (int i = 0; i < 4; ++i) {
            int f = t + i * 256, r = f >> 5, c4 = f & 31;
            Bs[buf][c4 * 4 + 0][r] = f2bf(rb[i].x);
            Bs[buf][c4 * 4 + 1][r] = f2bf(rb[i].y);
            Bs[buf][c4 * 4 + 2][r] = f2bf(rb[i].z);
            Bs[buf][c4 * 4 + 3][r] = f2bf(rb[i].w);
        }
    };

    v8f acc[2][4] = {};

    ld_regs(0);
    st_tile(0);
    __syncthreads();

    int cur = 0;
    for (int k0 = 0; k0 < K; k0 += BK) {
        const bool more = (k0 + BK) < K;
        if (more) ld_regs(k0 + BK);                 // issue next-tile global loads

        v16bf af[2], bfr[4];
        af[0] = frag_ld(&As[cur][wm + l16][0], kg, 0);
        af[1] = frag_ld(&As[cur][wm + 16 + l16][0], kg, 0);
#pragma unroll
        for (int ni = 0; ni < 4; ++ni)
            bfr[ni] = frag_ld(&Bs[cur][wn + ni * 16 + l16][0], kg, 0);
#pragma unroll
        for (int mi = 0; mi < 2; ++mi)
#pragma unroll
            for (int ni = 0; ni < 4; ++ni)
                acc[mi][ni] = __builtin_amdgcn_wmma_f32_16x16x32_bf16(
                    false, af[mi], false, bfr[ni], (short)0, acc[mi][ni], false, false);

        if (more) st_tile(cur ^ 1);
        __syncthreads();
        cur ^= 1;
    }

    // ---- epilogue ----
#pragma unroll
    for (int mi = 0; mi < 2; ++mi) {
#pragma unroll
        for (int ni = 0; ni < 4; ++ni) {
#pragma unroll
            for (int e = 0; e < 8; ++e) {
                int gm = m0 + wm + mi * 16 + kg * 8 + e;   // M = (lane>>4)*8+e
                int gn = n0 + wn + ni * 16 + l16;          // N = lane&15
                float v = acc[mi][ni][e];
                if constexpr (EPI == 0) {
                    int b = gm >> 11, tt = gm & 2047;
                    int which = gn >> 11;
                    int r = gn & 2047, h = r >> 7, d = r & 127;
                    bf16_t bv = f2bf(v);
                    if (which == 0)
                        qb[(((size_t)b * 16 + h) * 2048 + tt) * 128 + d] = bv;
                    else if (which == 1)
                        kb[(((size_t)b * 16 + h) * 2048 + tt) * 128 + d] = bv;
                    else  // V stored transposed: [B,H,hd,T]
                        vb[(((size_t)b * 16 + h) * 128 + d) * 2048 + tt] = bv;
                } else {
                    outp[(size_t)gm * N + gn] = v;
                }
            }
        }
    }
}

// ---------------------------------------------------------------------------
// RoPE in-place on q,k buffers laid out [B*H, T, 128] (bf16).
// ---------------------------------------------------------------------------
__global__ __launch_bounds__(256)
void rope_kernel(bf16_t* __restrict__ qb, bf16_t* __restrict__ kb) {
    int p = blockIdx.x * blockDim.x + threadIdx.x;     // < 32*2048*64 = 2^22
    int d2 = p & 63;
    int tp = (p >> 6) & 2047;
    int bh = p >> 17;
    float freq = __powf(10000.0f, -((float)(2 * d2)) * (1.0f / 128.0f));
    float ang = (float)tp * freq;
    float s, c;
    __sincosf(ang, &s, &c);
    size_t base = (((size_t)bh * 2048 + tp) * 128) + 2 * d2;
    float q0 = bf2f(qb[base]), q1 = bf2f(qb[base + 1]);
    qb[base]     = f2bf(q0 * c - q1 * s);
    qb[base + 1] = f2bf(q1 * c + q0 * s);
    float k0 = bf2f(kb[base]), k1 = bf2f(kb[base + 1]);
    kb[base]     = f2bf(k0 * c - k1 * s);
    kb[base + 1] = f2bf(k1 * c + k0 * s);
}

// ---------------------------------------------------------------------------
// Flash attention: per block one (b,h) and one 64-row q tile.
// 128 threads = 4 waves; each wave owns 16 q rows, full hd=128 accumulator.
// K in [T,hd]; V in TRANSPOSED [hd,T] so both WMMA operand streams are
// contiguous ds_load_b128. Double-buffered K/V tiles with async LDS copies.
// ---------------------------------------------------------------------------
__global__ __launch_bounds__(128)
void attn_kernel(const bf16_t* __restrict__ qb, const bf16_t* __restrict__ kbuf,
                 const bf16_t* __restrict__ vbuf, bf16_t* __restrict__ yb) {
    constexpr int TSEQ = 2048, HD = 128, KT = 64;
    const int bh = blockIdx.x;            // 0..31
    const int q0 = blockIdx.y * 64;       // q tile start
    const int t = threadIdx.x;
    const int wave = t >> 5, lane = t & 31, l16 = lane & 15, kg = lane >> 4;

    const bf16_t* Q  = qb   + (size_t)bh * TSEQ * HD;
    const bf16_t* Kp = kbuf + (size_t)bh * TSEQ * HD;
    const bf16_t* Vt = vbuf + (size_t)bh * HD * TSEQ;   // [hd][T]

    __shared__ alignas(16) bf16_t Ks[2][KT][HD];   // [key][d]
    __shared__ alignas(16) bf16_t Vs[2][HD][KT];   // [d][key]
    __shared__ alignas(16) bf16_t Ps[4][16][KT];

    // tile copy: K tile is contiguous 16KB; V tile is 128 rows of 128B
    auto tile_cp = [&](int buf, int kt) {
        const uint4* gk = (const uint4*)(Kp + (size_t)kt * HD);
        uint4* sk = (uint4*)&Ks[buf][0][0];
        uint4* sv = (uint4*)&Vs[buf][0][0];
#pragma unroll
        for (int i = 0; i < 8; ++i) {
            int c = t + i * 128;
            cp16_g2l(gk + c, sk + c);
            int d = c >> 3, c8 = c & 7;                 // 8 uint4 per V row
            cp16_g2l((const uint4*)(Vt + (size_t)d * TSEQ + kt) + c8, sv + c);
        }
    };

    // Q fragments for this wave's 16 rows (hd=128 -> 4 K-steps of 32)
    const int qrow = q0 + wave * 16 + l16;
    v16bf qf[4];
#pragma unroll
    for (int ks = 0; ks < 4; ++ks)
        qf[ks] = frag_ld(Q + (size_t)qrow * HD, kg, ks * 32);

    v8f o[8] = {};
    float mrow[8], lrow[8];
#pragma unroll
    for (int e = 0; e < 8; ++e) { mrow[e] = -3.0e38f; lrow[e] = 0.0f; }
    const float scale = 0.08838834764831845f;   // 1/sqrt(128)

    tile_cp(0, 0);
    cp_wait();
    __syncthreads();

    int cur = 0;
    for (int kt = 0; kt < TSEQ; kt += KT) {
        if (kt + KT < TSEQ) tile_cp(cur ^ 1, kt + KT);   // prefetch next tile

        // S = Q * K^T  (16 x 64 per wave; contraction over hd=128)
        v8f sacc[4] = {};
#pragma unroll
        for (int ni = 0; ni < 4; ++ni) {
#pragma unroll
            for (int ks = 0; ks < 4; ++ks) {
                v16bf bf = frag_ld(&Ks[cur][ni * 16 + l16][0], kg, ks * 32);
                sacc[ni] = __builtin_amdgcn_wmma_f32_16x16x32_bf16(
                    false, qf[ks], false, bf, (short)0, sacc[ni], false, false);
            }
        }
#pragma unroll
        for (int ni = 0; ni < 4; ++ni)
#pragma unroll
            for (int e = 0; e < 8; ++e) sacc[ni][e] *= scale;

        // online softmax (row = kg*8+e, spread across 16 lanes)
        float alpha[8];
#pragma unroll
        for (int e = 0; e < 8; ++e) {
            float vm = fmaxf(fmaxf(sacc[0][e], sacc[1][e]),
                             fmaxf(sacc[2][e], sacc[3][e]));
            vm = half_reduce_max(vm);
            float nm = fmaxf(mrow[e], vm);
            alpha[e] = __expf(mrow[e] - nm);
            mrow[e] = nm;
            lrow[e] *= alpha[e];
        }
#pragma unroll
        for (int no = 0; no < 8; ++no)
#pragma unroll
            for (int e = 0; e < 8; ++e) o[no][e] *= alpha[e];

#pragma unroll
        for (int e = 0; e < 8; ++e) {
            float rs = 0.0f;
#pragma unroll
            for (int ni = 0; ni < 4; ++ni) {
                float p = __expf(sacc[ni][e] - mrow[e]);
                sacc[ni][e] = p;
                rs += p;
            }
            lrow[e] += half_reduce_sum(rs);
        }

        // stage P (bf16) into per-wave LDS, re-read as A fragments
#pragma unroll
        for (int ni = 0; ni < 4; ++ni)
#pragma unroll
            for (int e = 0; e < 8; ++e)
                Ps[wave][kg * 8 + e][ni * 16 + l16] = f2bf(sacc[ni][e]);

        // O += P * V   (contraction over 64 keys = 2 K-steps, V is [d][key])
#pragma unroll
        for (int ks2 = 0; ks2 < 2; ++ks2) {
            v16bf pa = frag_ld(&Ps[wave][l16][0], kg, ks2 * 32);
#pragma unroll
            for (int no = 0; no < 8; ++no) {
                v16bf vf = frag_ld(&Vs[cur][no * 16 + l16][0], kg, ks2 * 32);
                o[no] = __builtin_amdgcn_wmma_f32_16x16x32_bf16(
                    false, pa, false, vf, (short)0, o[no], false, false);
            }
        }

        cp_wait();
        __syncthreads();
        cur ^= 1;
    }

    // normalize and write y (bf16, row-major [B*T, 2048])
    const int b = bh >> 4, h = bh & 15;
#pragma unroll
    for (int no = 0; no < 8; ++no) {
#pragma unroll
        for (int e = 0; e < 8; ++e) {
            float v = o[no][e] / lrow[e];
            int grow = b * TSEQ + q0 + wave * 16 + kg * 8 + e;
            int gcol = h * HD + no * 16 + l16;
            yb[(size_t)grow * 2048 + gcol] = f2bf(v);
        }
    }
}

// ---------------------------------------------------------------------------
// Launch
// ---------------------------------------------------------------------------
extern "C" void kernel_launch(void* const* d_in, const int* in_sizes, int n_in,
                              void* d_out, int out_size, void* d_ws, size_t ws_size,
                              hipStream_t stream) {
    const float* x      = (const float*)d_in[0];   // [2,2048,2048]
    const float* w_attn = (const float*)d_in[1];   // [2048,6144]
    const float* w_proj = (const float*)d_in[2];   // [2048,2048]
    float* out = (float*)d_out;                    // [2,2048,2048]

    const size_t NELEM = (size_t)2 * 16 * 2048 * 128;   // 8,388,608 per buffer
    bf16_t* qb = (bf16_t*)d_ws;
    bf16_t* kb = qb + NELEM;
    bf16_t* vb = kb + NELEM;                       // transposed [B,H,hd,T]
    bf16_t* yb = vb + NELEM;                       // [4096,2048] bf16

    // 1) qkv = x @ w_attn, scatter to q/k [B,H,T,hd], v [B,H,hd,T] (bf16)
    gemm_wmma<0, false><<<dim3(6144 / 128, 4096 / 128), 256, 0, stream>>>(
        (const void*)x, w_attn, 4096, 6144, 2048, qb, kb, vb, nullptr);

    // 2) RoPE in place on q,k
    rope_kernel<<<16384, 256, 0, stream>>>(qb, kb);

    // 3) flash attention -> y
    attn_kernel<<<dim3(32, 2048 / 64), 128, 0, stream>>>(qb, kb, vb, yb);

    // 4) out = y @ w_proj (fp32 store)
    gemm_wmma<1, true><<<dim3(2048 / 128, 4096 / 128), 256, 0, stream>>>(
        (const void*)yb, w_proj, 4096, 2048, 2048, nullptr, nullptr, nullptr, out);
}
